// GraphHead_32770600469192
// MI455X (gfx1250) — matrix operand order repaired and checked by reading, not compiled, not run on hardware
//
#include <hip/hip_runtime.h>
#include <hip/hip_bf16.h>
#include <stdint.h>

// ---------------------------------------------------------------------------
// Types for CDNA5 WMMA (gfx1250, wave32): v_wmma_f32_16x16x32_bf16
// ---------------------------------------------------------------------------
typedef __attribute__((ext_vector_type(16))) __bf16 v16bf;
typedef __attribute__((ext_vector_type(2)))  __bf16 v2bf;
typedef __attribute__((ext_vector_type(8)))  float  v8f;

union FragBF {
  unsigned int u[8];
  v16bf v;
};

__device__ __forceinline__ unsigned int rne_bump(float f) {
  union { float f; unsigned int u; } x; x.f = f;
  return x.u + 0x7FFFu + ((x.u >> 16) & 1u);   // bf16 RNE: result in high 16 bits
}

// Pack two fp32 -> two bf16 (RNE) in one u32.
//   v_bfe+v_add3 per float, then one v_perm_b32 byte-select of the high halves.
__device__ __forceinline__ unsigned int pack_bf2(float lo, float hi) {
#if __has_builtin(__builtin_amdgcn_cvt_pk_bf16_f32)
  union { v2bf v; unsigned int u; } r;
  r.v = __builtin_amdgcn_cvt_pk_bf16_f32(lo, hi);
  return r.u;
#elif __has_builtin(__builtin_amdgcn_perm)
  // result bytes {3,2,1,0} = {a.b3, a.b2, b.b3, b.b2}  (a=hi, b=lo)
  return __builtin_amdgcn_perm(rne_bump(hi), rne_bump(lo), 0x07060302u);
#else
  return ((rne_bump(lo) >> 16) & 0xFFFFu) | (rne_bump(hi) & 0xFFFF0000u);
#endif
}

#define BM 128
#define BN 128
#define BK 32
#define BKP 40   // padded LDS row stride (elements); 80B rows keep 16B alignment

// ---------------------------------------------------------------------------
// Generic fused GEMM:  C = act( (A@B + bias) * (mulA[rA]+mulB[rB]+mulbias) )
//   A: MxK fp32 (lda), B: KxN fp32, C: MxN fp32 (ldc)
//   bperm==1: B[k][n] = Bptr[(n>>6)*bKtot*64 + k*64 + (n&63)]  (W[c,i,s] layout)
//   gate row indices: rA=(row/divA)%modA, rB=(row/divB)%modB (mul* width == N)
// Block = 256 threads = 8 wave32s; block tile 128x128; wave tile 32x64.
// GUARD=false requires M%128==0 && K%32==0 (all heavy GEMMs) -> branchless
// vectorized staging. GUARD=true uses clamped (branchless) scalar staging.
// ---------------------------------------------------------------------------
template <bool GUARD>
__global__ __launch_bounds__(256) void wmma_gemm_kernel(
    const float* __restrict__ A, int lda,
    const float* __restrict__ B, int ldb, int bperm, int bKtot,
    float* __restrict__ C, int ldc,
    const float* __restrict__ bias,
    const float* __restrict__ mulA, int divA, int modA,
    const float* __restrict__ mulB, int divB, int modB,
    const float* __restrict__ mulbias,
    int M, int N, int K, int do_relu)
{
  __shared__ unsigned short As[BM][BKP];
  __shared__ unsigned short Bs[BN][BKP];   // stored transposed: Bs[n][k]

  const int tid  = threadIdx.x;
  const int wave = tid >> 5;
  const int lane = tid & 31;
  const int half = lane >> 4;   // lane 0-15 vs 16-31
  const int l15  = lane & 15;
  const int wm   = wave >> 1;   // 0..3 -> M offset wm*32
  const int wn   = wave & 1;    // 0..1 -> N offset wn*64
  const int bMo  = blockIdx.y * BM;
  const int bNo  = blockIdx.x * BN;

  v8f acc[2][4];
  #pragma unroll
  for (int i = 0; i < 2; ++i)
    #pragma unroll
    for (int j = 0; j < 4; ++j)
      #pragma unroll
      for (int e = 0; e < 8; ++e) acc[i][j][e] = 0.0f;

  // Per-thread staging coordinates (fixed across K-steps)
  const int ar  = tid >> 1;              // A: row 0..127
  const int ac4 = (tid & 1) << 4;        // A: col {0,16} -> 2 float4 each
  const int bn  = tid >> 1;              // B: n 0..127
  const int bk4 = (tid & 1) << 4;        // B: k {0,16} -> 2 groups of 4

  for (int k0 = 0; k0 < K; k0 += BK) {
    if constexpr (!GUARD) {
      // ---- A tile: 128x32, float4 loads, packed bf16 ds_store_b64 ----
      #pragma unroll
      for (int g = 0; g < 2; ++g) {            // two float4s per thread
        int c = ac4 + g * 4;
        const float4 v = *(const float4*)&A[(size_t)(bMo + ar) * (size_t)lda + k0 + c];
        uint2 p; p.x = pack_bf2(v.x, v.y); p.y = pack_bf2(v.z, v.w);
        *(uint2*)&As[ar][c] = p;
      }
      // ---- B tile: 32x128 -> Bs[n][k], 4 coalesced b32 loads per group ----
      #pragma unroll
      for (int g = 0; g < 2; ++g) {
        int kbase = bk4 + g * 4;
        int n = bNo + bn;
        float t0, t1, t2, t3;
        if (bperm) {
          const float* bp = &B[(size_t)(n >> 6) * (size_t)bKtot * 64 +
                               (size_t)(k0 + kbase) * 64 + (size_t)(n & 63)];
          t0 = bp[0]; t1 = bp[64]; t2 = bp[128]; t3 = bp[192];
        } else {
          const float* bp = &B[(size_t)(k0 + kbase) * (size_t)ldb + (size_t)n];
          t0 = bp[0]; t1 = bp[(size_t)ldb]; t2 = bp[(size_t)2 * ldb]; t3 = bp[(size_t)3 * ldb];
        }
        uint2 p; p.x = pack_bf2(t0, t1); p.y = pack_bf2(t2, t3);
        *(uint2*)&Bs[bn][kbase] = p;
      }
    } else {
      // ---- guarded (branchless clamped) staging ----
      #pragma unroll
      for (int it = 0; it < 4; ++it) {
        int i  = tid + it * 256;
        int r  = i >> 3;
        int c4 = (i & 7) << 2;
        int grow = bMo + r;
        int rc = grow < M ? grow : (M - 1);
        #pragma unroll
        for (int j = 0; j < 4; j += 2) {
          int kg0 = k0 + c4 + j;
          int kc0 = kg0 < K ? kg0 : (K - 1);
          int kc1 = (kg0 + 1) < K ? (kg0 + 1) : (K - 1);
          float v0 = A[(size_t)rc * (size_t)lda + kc0];
          float v1 = A[(size_t)rc * (size_t)lda + kc1];
          v0 = (grow < M && kg0 < K) ? v0 : 0.0f;
          v1 = (grow < M && (kg0 + 1) < K) ? v1 : 0.0f;
          *(unsigned int*)&As[r][c4 + j] = pack_bf2(v0, v1);
        }
      }
      #pragma unroll
      for (int it = 0; it < 4; ++it) {
        int i  = tid + it * 256;
        int kk = i >> 5;
        int n4 = (i & 31) << 2;
        int kg = k0 + kk;
        int kc = kg < K ? kg : (K - 1);
        #pragma unroll
        for (int j = 0; j < 4; ++j) {
          int n = bNo + n4 + j;
          size_t offc = bperm
              ? ((size_t)(n >> 6) * (size_t)bKtot * 64 + (size_t)kc * 64 + (size_t)(n & 63))
              : ((size_t)kc * (size_t)ldb + (size_t)n);
          float v = B[offc];
          v = (kg < K) ? v : 0.0f;
          Bs[n4 + j][kk] = (unsigned short)(pack_bf2(v, 0.0f) & 0xFFFFu);
        }
      }
    }
    __syncthreads();

    // ---- build fragments: k = half*16 + e for both A and B ----
    // A (16x32): lane m = l15; VGPR0-3 hold K = half*16 + 0..7, VGPR4-7 K = half*16 + 8..15
    FragBF fa[2];
    #pragma unroll
    for (int mt = 0; mt < 2; ++mt) {
      const unsigned int* p0 =
          (const unsigned int*)&As[wm * 32 + mt * 16 + l15][half * 16];
      const unsigned int* p1 =
          (const unsigned int*)&As[wm * 32 + mt * 16 + l15][half * 16 + 8];
      #pragma unroll
      for (int j = 0; j < 4; ++j) { fa[mt].u[j] = p0[j]; fa[mt].u[4 + j] = p1[j]; }
    }
    // B (32x16): lane n = l15; elements e -> K = half*16 + e (contiguous 32B)
    #pragma unroll
    for (int nt = 0; nt < 4; ++nt) {
      FragBF fb;
      const unsigned int* p =
          (const unsigned int*)&Bs[wn * 64 + nt * 16 + l15][half * 16];
      #pragma unroll
      for (int j = 0; j < 8; ++j) fb.u[j] = p[j];
      #pragma unroll
      for (int mt = 0; mt < 2; ++mt) {
        acc[mt][nt] = __builtin_amdgcn_wmma_f32_16x16x32_bf16(
            false, fa[mt].v, false, fb.v, (short)0, acc[mt][nt], false, false);
      }
    }
    __syncthreads();
  }

  // ---- epilogue: C/D layout -> row = VGPRidx + half*8 ----
  #pragma unroll
  for (int mt = 0; mt < 2; ++mt) {
    #pragma unroll
    for (int nt = 0; nt < 4; ++nt) {
      int col = bNo + wn * 64 + nt * 16 + l15;
      #pragma unroll
      for (int v = 0; v < 8; ++v) {
        int row = bMo + wm * 32 + mt * 16 + half * 8 + v;
        if (GUARD ? (row < M) : true) {
          float x = acc[mt][nt][v];
          if (bias) x += bias[col];
          if (mulA) {
            float f = mulA[(size_t)((row / divA) % modA) * (size_t)N + col];
            if (mulB)    f += mulB[(size_t)((row / divB) % modB) * (size_t)N + col];
            if (mulbias) f += mulbias[col];
            x *= f;
          }
          if (do_relu) x = fmaxf(x, 0.0f);
          C[(size_t)row * (size_t)ldc + col] = x;
        }
      }
    }
  }
}

// ---------------------------------------------------------------------------
// Small helper kernels
// ---------------------------------------------------------------------------
__global__ void sum16_kernel(const float* __restrict__ b3, float* __restrict__ out, int n) {
  int r = blockIdx.x * blockDim.x + threadIdx.x;
  if (r < n) {
    float s = 0.f;
    for (int c = 0; c < 16; ++c) s += b3[(size_t)c * n + r];
    out[r] = s;
  }
}

// adj[p] = dot(W[p,:], adjW) + adjb   (one block per row p)
__global__ void dot_rows_kernel(const float* __restrict__ Wmat, const float* __restrict__ w,
                                const float* __restrict__ b, float* __restrict__ out, int cols) {
  __shared__ float red[256];
  int p = blockIdx.x;
  float s = 0.f;
  for (int r = threadIdx.x; r < cols; r += 256)
    s += Wmat[(size_t)p * cols + r] * w[r];
  red[threadIdx.x] = s;
  __syncthreads();
  for (int off = 128; off > 0; off >>= 1) {
    if (threadIdx.x < off) red[threadIdx.x] += red[threadIdx.x + off];
    __syncthreads();
  }
  if (threadIdx.x == 0) out[p] = red[0] + b[0];
}

// softmax over `cols` (<=256) elements, strided input, contiguous output
__global__ void softmax_kernel(const float* __restrict__ in, float* __restrict__ out,
                               int cols, int rstride, int cstride) {
  __shared__ float red[256];
  int row = blockIdx.x;
  int t = threadIdx.x;
  float v = (t < cols) ? in[(size_t)row * rstride + (size_t)t * cstride] : -3.4e38f;
  red[t] = v; __syncthreads();
  for (int off = 128; off > 0; off >>= 1) {
    if (t < off) red[t] = fmaxf(red[t], red[t + off]);
    __syncthreads();
  }
  float mx = red[0]; __syncthreads();
  float e = (t < cols) ? __expf(v - mx) : 0.f;
  red[t] = e; __syncthreads();
  for (int off = 128; off > 0; off >>= 1) {
    if (t < off) red[t] += red[t + off];
    __syncthreads();
  }
  float s = red[0];
  if (t < cols) out[(size_t)row * cols + t] = e / s;
}

// out[b,r] = relu( sum_i soft[b,i] * M[(b*pbase + i*pstride), r] )   r in [0,1024)
__global__ void msg_kernel(const float* __restrict__ Mmat, const float* __restrict__ soft,
                           float* __restrict__ out, int count, int pbase, int pstride,
                           int softld) {
  int b = blockIdx.x;
  for (int r = threadIdx.x; r < 1024; r += 256) {
    float s = 0.f;
    for (int i = 0; i < count; ++i) {
      s += soft[(size_t)b * softld + i] *
           Mmat[((size_t)(b * pbase + i * pstride)) * 1024 + r];
    }
    out[(size_t)b * 1024 + r] = fmaxf(s, 0.f);
  }
}

// out[row,:] = LN(base[row,:] + msg[row,:]) * g + b  over 1024 features
__global__ void layernorm_kernel(const float* __restrict__ base, const float* __restrict__ msg,
                                 const float* __restrict__ g, const float* __restrict__ bb,
                                 float* __restrict__ out) {
  __shared__ float r1[256], r2[256];
  int row = blockIdx.x;
  int t = threadIdx.x;
  float s = 0.f, s2 = 0.f;
  for (int c = t; c < 1024; c += 256) {
    float x = base[(size_t)row * 1024 + c] + msg[(size_t)row * 1024 + c];
    s += x; s2 += x * x;
  }
  r1[t] = s; r2[t] = s2; __syncthreads();
  for (int off = 128; off > 0; off >>= 1) {
    if (t < off) { r1[t] += r1[t + off]; r2[t] += r2[t + off]; }
    __syncthreads();
  }
  float mu  = r1[0] * (1.f / 1024.f);
  float var = r2[0] * (1.f / 1024.f) - mu * mu;
  float inv = rsqrtf(var + 1e-5f);
  for (int c = t; c < 1024; c += 256) {
    float x = base[(size_t)row * 1024 + c] + msg[(size_t)row * 1024 + c];
    out[(size_t)row * 1024 + c] = (x - mu) * inv * g[c] + bb[c];
  }
}

// ---------------------------------------------------------------------------
// Host-side orchestration
// ---------------------------------------------------------------------------
static inline void launch_gemm(hipStream_t s, const float* A, int lda,
                               const float* B, int ldb, int bperm, int bKtot,
                               float* C, int ldc, const float* bias,
                               const float* mulA, int divA, int modA,
                               const float* mulB, int divB, int modB,
                               const float* mulbias, int M, int N, int K, int relu) {
  dim3 grid(N / BN, (M + BM - 1) / BM);
  if ((M % BM == 0) && (K % BK == 0)) {
    wmma_gemm_kernel<false><<<grid, 256, 0, s>>>(A, lda, B, ldb, bperm, bKtot, C, ldc,
        bias, mulA, divA, modA, mulB, divB, modB, mulbias, M, N, K, relu);
  } else {
    wmma_gemm_kernel<true><<<grid, 256, 0, s>>>(A, lda, B, ldb, bperm, bKtot, C, ldc,
        bias, mulA, divA, modA, mulB, divB, modB, mulbias, M, N, K, relu);
  }
}

extern "C" void kernel_launch(void* const* d_in, const int* in_sizes, int n_in,
                              void* d_out, int out_size, void* d_ws, size_t ws_size,
                              hipStream_t stream) {
  (void)in_sizes; (void)n_in; (void)out_size; (void)ws_size;
  const float* node   = (const float*)d_in[0];   // (256,1024)
  const float* spat   = (const float*)d_in[1];   // (16384,36)
  const float* gfeat  = (const float*)d_in[2];   // (1,256)
  const float* sp_W1  = (const float*)d_in[3];
  const float* sp_b1  = (const float*)d_in[4];
  const float* sp_W2  = (const float*)d_in[5];
  const float* sp_b2  = (const float*)d_in[6];
  const float* sp_W3  = (const float*)d_in[7];
  const float* sp_b3  = (const float*)d_in[8];
  const float* ah_W1  = (const float*)d_in[9];   // (16,2048,64)
  const float* ah_b1  = (const float*)d_in[10];  // (16,64) -> flat 1024
  const float* ah_W2  = (const float*)d_in[11];  // (16,1024,64)
  const float* ah_b2  = (const float*)d_in[12];
  const float* ah_W3  = (const float*)d_in[13];  // (16,64,1024) == flat (1024,1024)
  const float* ah_b3  = (const float*)d_in[14];  // (16,1024)
  const float* o2s_W1 = (const float*)d_in[15];
  const float* o2s_b1 = (const float*)d_in[16];
  const float* o2s_W2 = (const float*)d_in[17];
  const float* o2s_b2 = (const float*)d_in[18];
  const float* o2s_W3 = (const float*)d_in[19];
  const float* o2s_b3 = (const float*)d_in[20];
  const float* s2o_W1 = (const float*)d_in[21];
  const float* s2o_b1 = (const float*)d_in[22];
  const float* s2o_W2 = (const float*)d_in[23];
  const float* s2o_b2 = (const float*)d_in[24];
  const float* s2o_W3 = (const float*)d_in[25];
  const float* s2o_b3 = (const float*)d_in[26];
  const float* ahg_W1 = (const float*)d_in[27];  // (16,256,64)
  const float* ahg_b1 = (const float*)d_in[28];
  const float* ahg_W2 = (const float*)d_in[29];
  const float* ahg_b2 = (const float*)d_in[30];
  const float* ahg_W3 = (const float*)d_in[31];
  const float* ahg_b3 = (const float*)d_in[32];
  const float* adj_W  = (const float*)d_in[33];  // (1024,1)
  const float* adj_b  = (const float*)d_in[34];
  const float* lnh_g  = (const float*)d_in[35];
  const float* lnh_b  = (const float*)d_in[36];
  const float* lno_g  = (const float*)d_in[37];
  const float* lno_b  = (const float*)d_in[38];

  float* out = (float*)d_out;                    // (16384, 2048)

  // ---- workspace carve-up ----
  float* ws = (float*)d_ws;
  size_t off = 0;
  auto alloc = [&](size_t n) { float* p = ws + off; off += n; return p; };
  const size_t BIG = (size_t)16384 * 1024;
  float* SP    = alloc(BIG);        // sp (P,1024)
  float* BUF1  = alloc(BIG);
  float* BUF2  = alloc(BIG);
  float* T1    = alloc(64 * 1024);
  float* T2    = alloc(256 * 1024);
  float* A1o   = alloc(256 * 1024);
  float* A1s   = alloc(64 * 1024);
  float* a1g   = alloc(1024);
  float* sb_ah = alloc(1024);
  float* sb_o2 = alloc(1024);
  float* sb_s2 = alloc(1024);
  float* sb_ag = alloc(1024);
  float* adjv  = alloc(16384);
  float* sof_h = alloc(16384);      // (64,256)
  float* sof_o = alloc(16384);      // (256,64)
  float* msg_h = alloc(64 * 1024);
  float* msg_o = alloc(256 * 1024);
  float* henc2 = alloc(64 * 1024);
  float* oenc2 = alloc(256 * 1024);

  const float* NUL = nullptr;

  // sum_c b3 -> (1024,) for each pair-head
  sum16_kernel<<<4, 256, 0, stream>>>(ah_b3,  sb_ah, 1024);
  sum16_kernel<<<4, 256, 0, stream>>>(o2s_b3, sb_o2, 1024);
  sum16_kernel<<<4, 256, 0, stream>>>(s2o_b3, sb_s2, 1024);
  sum16_kernel<<<4, 256, 0, stream>>>(ahg_b3, sb_ag, 1024);

  // ---- sp MLP: 16384x36 -> 128 -> 256 -> 1024 (relu each) ----
  launch_gemm(stream, spat, 36,  sp_W1, 128, 0, 0, BUF2, 128, sp_b1,
              NUL,1,1, NUL,1,1, NUL, 16384, 128, 36, 1);
  launch_gemm(stream, BUF2, 128, sp_W2, 256, 0, 0, BUF1, 256, sp_b2,
              NUL,1,1, NUL,1,1, NUL, 16384, 256, 128, 1);
  launch_gemm(stream, BUF1, 256, sp_W3, 1024, 0, 0, SP, 1024, sp_b3,
              NUL,1,1, NUL,1,1, NUL, 16384, 1024, 256, 1);

  // ---- pair_head #1 (weights): a1 split into T1 (h_enc) + T2 (o_enc) ----
  launch_gemm(stream, node, 1024, ah_W1, 0, 1, 2048, T1, 1024, NUL,
              NUL,1,1, NUL,1,1, NUL, 64, 1024, 1024, 0);
  launch_gemm(stream, node, 1024, ah_W1 + (size_t)1024 * 64, 0, 1, 2048, T2, 1024, NUL,
              NUL,1,1, NUL,1,1, NUL, 256, 1024, 1024, 0);
  // H = relu((SP@W2 + b2) * (T1[p/256] + T2[p%256] + b1))
  launch_gemm(stream, SP, 1024, ah_W2, 0, 1, 1024, BUF1, 1024, ah_b2,
              T1, 256, 64, T2, 1, 256, ah_b1, 16384, 1024, 1024, 1);
  // weights = relu(H @ W3flat + sum_c b3)
  launch_gemm(stream, BUF1, 1024, ah_W3, 1024, 0, 0, BUF2, 1024, sb_ah,
              NUL,1,1, NUL,1,1, NUL, 16384, 1024, 1024, 1);

  // ---- adj + softmaxes ----
  dot_rows_kernel<<<16384, 256, 0, stream>>>(BUF2, adj_W, adj_b, adjv, 1024);
  softmax_kernel<<<64,  256, 0, stream>>>(adjv, sof_h, 256, 256, 1);   // rows h
  softmax_kernel<<<256, 256, 0, stream>>>(adjv, sof_o, 64,  1,  256);  // rows n (adj.T)

  // ---- o2s: msg to h_enc ----
  launch_gemm(stream, node, 1024, o2s_W1, 0, 1, 1024, A1o, 1024, o2s_b1,
              NUL,1,1, NUL,1,1, NUL, 256, 1024, 1024, 0);
  launch_gemm(stream, SP, 1024, o2s_W2, 0, 1, 1024, BUF1, 1024, o2s_b2,
              A1o, 1, 256, NUL,1,1, NUL, 16384, 1024, 1024, 1);
  launch_gemm(stream, BUF1, 1024, o2s_W3, 1024, 0, 0, BUF2, 1024, sb_o2,
              NUL,1,1, NUL,1,1, NUL, 16384, 1024, 1024, 0);
  msg_kernel<<<64, 256, 0, stream>>>(BUF2, sof_h, msg_h, 256, 256, 1, 256);
  layernorm_kernel<<<64, 256, 0, stream>>>(node, msg_h, lnh_g, lnh_b, henc2);

  // ---- s2o: msg to o_enc ----
  launch_gemm(stream, henc2, 1024, s2o_W1, 0, 1, 1024, A1s, 1024, s2o_b1,
              NUL,1,1, NUL,1,1, NUL, 64, 1024, 1024, 0);
  launch_gemm(stream, SP, 1024, s2o_W2, 0, 1, 1024, BUF1, 1024, s2o_b2,
              A1s, 256, 64, NUL,1,1, NUL, 16384, 1024, 1024, 1);
  launch_gemm(stream, BUF1, 1024, s2o_W3, 1024, 0, 0, BUF2, 1024, sb_s2,
              NUL,1,1, NUL,1,1, NUL, 16384, 1024, 1024, 0);
  msg_kernel<<<256, 256, 0, stream>>>(BUF2, sof_o, msg_o, 64, 1, 256, 64);
  layernorm_kernel<<<256, 256, 0, stream>>>(node, msg_o, lno_g, lno_b, oenc2);

  // ---- pair_head #2 (f_local) with updated encodings -> out[:, 0:1024] ----
  launch_gemm(stream, henc2, 1024, ah_W1, 0, 1, 2048, T1, 1024, NUL,
              NUL,1,1, NUL,1,1, NUL, 64, 1024, 1024, 0);
  launch_gemm(stream, oenc2, 1024, ah_W1 + (size_t)1024 * 64, 0, 1, 2048, T2, 1024, NUL,
              NUL,1,1, NUL,1,1, NUL, 256, 1024, 1024, 0);
  launch_gemm(stream, SP, 1024, ah_W2, 0, 1, 1024, BUF1, 1024, ah_b2,
              T1, 256, 64, T2, 1, 256, ah_b1, 16384, 1024, 1024, 1);
  launch_gemm(stream, BUF1, 1024, ah_W3, 1024, 0, 0, out, 2048, sb_ah,
              NUL,1,1, NUL,1,1, NUL, 16384, 1024, 1024, 1);

  // ---- pair_head global (f_glob) -> out[:, 1024:2048] ----
  launch_gemm(stream, gfeat, 256, ahg_W1, 0, 1, 256, a1g, 1024, ahg_b1,
              NUL,1,1, NUL,1,1, NUL, 1, 1024, 256, 0);
  launch_gemm(stream, SP, 1024, ahg_W2, 0, 1, 1024, BUF1, 1024, ahg_b2,
              a1g, 1, 1, NUL,1,1, NUL, 16384, 1024, 1024, 1);
  launch_gemm(stream, BUF1, 1024, ahg_W3, 1024, 0, 0, out + 1024, 2048, sb_ag,
              NUL,1,1, NUL,1,1, NUL, 16384, 1024, 1024, 1);
}
